// TransformerBlock_24197845746097
// MI455X (gfx1250) — matrix operand order repaired and checked by reading, not compiled, not run on hardware
//
#include <hip/hip_runtime.h>
#include <cstddef>
#include <cstdint>

// ---------------------------------------------------------------------------
// Types for CDNA5 WMMA (wave32)
// ---------------------------------------------------------------------------
typedef __attribute__((ext_vector_type(16))) __bf16 bf16x16;
typedef __attribute__((ext_vector_type(8)))  float  f32x8;

union FragB16 { bf16x16 v; unsigned short h[16]; unsigned int u[8]; };
union AccF32  { f32x8 v; float f[8]; };

__device__ __forceinline__ unsigned short f2bf(float x) {
    unsigned int u = __builtin_bit_cast(unsigned int, x);
    u = (u + 0x7FFFu + ((u >> 16) & 1u)) >> 16;   // round-to-nearest-even
    return (unsigned short)u;
}

__device__ __forceinline__ float mish_act(float x) {
    float sp = (x > 20.0f) ? x : log1pf(expf(x));
    return x * tanhf(sp);
}

// CDNA5 async global->LDS copy (16 bytes / lane), tracked by ASYNCcnt.
__device__ __forceinline__ void async_copy_b128(const unsigned short* lds_dst,
                                                const unsigned short* gsrc) {
    unsigned           l = (unsigned)(size_t)lds_dst;            // LDS offset
    unsigned long long g = (unsigned long long)(size_t)gsrc;     // global addr
    asm volatile("global_load_async_to_lds_b128 %0, %1, off"
                 :: "v"(l), "v"(g) : "memory");
}
__device__ __forceinline__ void wait_async0() {
    asm volatile("s_wait_asynccnt 0" ::: "memory");
}
__device__ __forceinline__ void wait_async4() {
    asm volatile("s_wait_asynccnt 4" ::: "memory");   // allow newest 4 in flight
}

// ---------------------------------------------------------------------------
// fp32 -> bf16 transposing weight conversion: out[n*K + k] = bf16(in[k*N + n])
// 32x32 tiles through LDS, coalesced on both sides. 256 threads.
// ---------------------------------------------------------------------------
__global__ void cvt_transpose_bf16(const float* __restrict__ in,
                                   unsigned short* __restrict__ out,
                                   int K, int N) {
    __shared__ float t[32][33];
    const int k0 = blockIdx.y * 32, n0 = blockIdx.x * 32;
    const int tx = threadIdx.x & 31, ty = threadIdx.x >> 5;   // ty 0..7
    #pragma unroll
    for (int r = ty; r < 32; r += 8)
        t[r][tx] = in[(size_t)(k0 + r) * N + n0 + tx];
    __syncthreads();
    #pragma unroll
    for (int r = ty; r < 32; r += 8)
        out[(size_t)(n0 + r) * K + k0 + tx] = f2bf(t[tx][r]);
}

// ---------------------------------------------------------------------------
// LayerNorm over D=1024, one row per block (256 threads), bf16 output
// ---------------------------------------------------------------------------
__global__ void layernorm_bf16(const float* __restrict__ x,
                               const float* __restrict__ gamma,
                               const float* __restrict__ beta,
                               unsigned short* __restrict__ out) {
    const int row = blockIdx.x;
    const int tid = threadIdx.x;           // 256 threads, 4 elems each
    const float* xr = x + (size_t)row * 1024;

    float4 v = ((const float4*)xr)[tid];
    float s  = v.x + v.y + v.z + v.w;
    float ss = v.x * v.x + v.y * v.y + v.z * v.z + v.w * v.w;

    __shared__ float rs[256], rss[256];
    rs[tid] = s; rss[tid] = ss;
    __syncthreads();
    for (int off = 128; off > 0; off >>= 1) {
        if (tid < off) { rs[tid] += rs[tid + off]; rss[tid] += rss[tid + off]; }
        __syncthreads();
    }
    float mean = rs[0] * (1.0f / 1024.0f);
    float var  = rss[0] * (1.0f / 1024.0f) - mean * mean;
    float rstd = rsqrtf(var + 1e-5f);

    float4 g = ((const float4*)gamma)[tid];
    float4 b = ((const float4*)beta)[tid];
    unsigned short* o = out + (size_t)row * 1024 + tid * 4;
    o[0] = f2bf((v.x - mean) * rstd * g.x + b.x);
    o[1] = f2bf((v.y - mean) * rstd * g.y + b.y);
    o[2] = f2bf((v.z - mean) * rstd * g.z + b.z);
    o[3] = f2bf((v.w - mean) * rstd * g.w + b.w);
}

// ---------------------------------------------------------------------------
// bf16 WMMA GEMM:  out = act(A[MxK] @ Bt^T + bias)  (+ optional residual)
// Bt is the PRE-TRANSPOSED weight: Bt[n*K + k] == B[k*N + n].
// Block tile 64x64, 128 threads = 4 waves, each wave a 32x32 tile
// (2x2 v_wmma_f32_16x16x32_bf16 accumulators). K step 32.
// Tiles staged via global_load_async_to_lds_b128 with a 2-deep double
// buffer: tile t+1's DMA overlaps tile t's WMMA (ASYNCcnt in-order).
// act: 0 = none, 1 = mish.  vT (optional): transposed bf16 side-copy of
// columns [2048,3072) -> vT[(col-2048)*M + row]  (used for attention V^T).
// ---------------------------------------------------------------------------
__global__ void gemm_bf16_wmma(const unsigned short* __restrict__ A,
                               const unsigned short* __restrict__ Bt,
                               const float* __restrict__ bias,
                               const float* __restrict__ residual,
                               unsigned short* __restrict__ outBf,
                               float* __restrict__ outF,
                               unsigned short* __restrict__ vT,
                               int M, int N, int K, int act) {
    __shared__ __align__(16) unsigned short As[2][64 * 32];   // [buf][m][k]
    __shared__ __align__(16) unsigned short Bs[2][64 * 32];   // [buf][n][k]

    const int tid  = threadIdx.x;
    const int lane = tid & 31;
    const int wave = tid >> 5;                 // 0..3
    const int m0 = blockIdx.y * 64;
    const int n0 = blockIdx.x * 64;
    const int wm = (wave >> 1) * 32;           // wave row origin in tile
    const int wn = (wave & 1) * 32;            // wave col origin in tile
    const int l15  = lane & 15;
    const int hsel = (lane < 16) ? 0 : 1;

    AccF32 acc[2][2];
    for (int i = 0; i < 2; ++i)
        for (int j = 0; j < 2; ++j)
            acc[i][j].v = (f32x8){};

    // staging decomposition: 64 rows x 32 cols = 2048 elems; 128 threads,
    // each moves two 16B segments per matrix (rows sr, sr+32).
    const int sr = tid >> 2;                   // 0..31
    const int sc = (tid & 3) * 8;              // 0,8,16,24

    // prologue: stage tile 0 into buffer 0 (4 async b128 ops / lane)
    async_copy_b128(&As[0][sr * 32 + sc],        &A [(size_t)(m0 + sr)      * K + sc]);
    async_copy_b128(&As[0][(sr + 32) * 32 + sc], &A [(size_t)(m0 + sr + 32) * K + sc]);
    async_copy_b128(&Bs[0][sr * 32 + sc],        &Bt[(size_t)(n0 + sr)      * K + sc]);
    async_copy_b128(&Bs[0][(sr + 32) * 32 + sc], &Bt[(size_t)(n0 + sr + 32) * K + sc]);

    for (int k0 = 0; k0 < K; k0 += 32) {
        const int cur = (k0 >> 5) & 1;
        const bool more = (k0 + 32) < K;

        __syncthreads();                 // all waves finished reading buf[cur^1]
        if (more) {                      // overlap: DMA tile t+1 during compute t
            const int kn = k0 + 32;
            async_copy_b128(&As[cur ^ 1][sr * 32 + sc],        &A [(size_t)(m0 + sr)      * K + kn + sc]);
            async_copy_b128(&As[cur ^ 1][(sr + 32) * 32 + sc], &A [(size_t)(m0 + sr + 32) * K + kn + sc]);
            async_copy_b128(&Bs[cur ^ 1][sr * 32 + sc],        &Bt[(size_t)(n0 + sr)      * K + kn + sc]);
            async_copy_b128(&Bs[cur ^ 1][(sr + 32) * 32 + sc], &Bt[(size_t)(n0 + sr + 32) * K + kn + sc]);
            if (k0 + 64 < K)
                __builtin_prefetch(&A[(size_t)(m0 + sr) * K + k0 + 64 + sc], 0, 3);
            wait_async4();               // own tile-t copies done; 4 newest in flight
        } else {
            wait_async0();
        }
        __syncthreads();                 // every wave's tile-t copies landed

        FragB16 fa[2], fb[2];
        #pragma unroll
        for (int i = 0; i < 2; ++i) {
            const int m = wm + i * 16 + l15;
            #pragma unroll
            for (int v = 0; v < 8; ++v) {
                const int kk = ((v < 4) ? 0 : 16) + hsel * 8 + 2 * (v & 3);
                fa[i].u[v] = *(const unsigned int*)&As[cur][m * 32 + kk];
            }
        }
        #pragma unroll
        for (int j = 0; j < 2; ++j) {
            const int n = wn + j * 16 + l15;
            #pragma unroll
            for (int v = 0; v < 8; ++v) {
                const int kk = ((v < 4) ? 0 : 16) + hsel * 8 + 2 * (v & 3);
                fb[j].u[v] = *(const unsigned int*)&Bs[cur][n * 32 + kk];
            }
        }
        #pragma unroll
        for (int i = 0; i < 2; ++i)
            #pragma unroll
            for (int j = 0; j < 2; ++j)
                acc[i][j].v = __builtin_amdgcn_wmma_f32_16x16x32_bf16(
                    false, fa[i].v, false, fb[j].v,
                    (short)0, acc[i][j].v, false, false);
    }

    // epilogue: C/D layout — lane<16: col=l15, row=p; lane>=16: row=p+8
    #pragma unroll
    for (int i = 0; i < 2; ++i) {
        #pragma unroll
        for (int j = 0; j < 2; ++j) {
            const int gcol = n0 + wn + j * 16 + l15;
            const float bsv = bias[gcol];
            #pragma unroll
            for (int p = 0; p < 8; ++p) {
                const int grow = m0 + wm + i * 16 + p + hsel * 8;
                float val = acc[i][j].f[p] + bsv;
                if (act == 1) val = mish_act(val);
                const size_t idx = (size_t)grow * N + gcol;
                unsigned short bv = f2bf(val);
                if (outF)  outF[idx]  = val + (residual ? residual[idx] : 0.0f);
                if (outBf) outBf[idx] = bv;
                if (vT && gcol >= 2048)
                    vT[(size_t)(gcol - 2048) * M + grow] = bv;
            }
        }
    }
}

// ---------------------------------------------------------------------------
// Fused attention: one workgroup per (b, head, 16-row q tile).
// qkv layout: [token(2048)][3072] bf16, col = which*1024 + head*64 + d.
// vT layout : [hd(1024)][token(2048)] bf16  (V transposed).
// scores (16x1024 fp32) live in LDS; exact softmax; QK^T and PV via WMMA.
// dyn-LDS: 64KB scores + 32KB p(bf16) + 2KB reductions.
// ---------------------------------------------------------------------------
__global__ void attention_wmma(const unsigned short* __restrict__ qkv,
                               const unsigned short* __restrict__ vT,
                               const float* __restrict__ attn_bias,
                               unsigned short* __restrict__ ctx) {
    extern __shared__ char smem[];
    float*          scores = (float*)smem;                         // 16*1024 f32
    unsigned short* pbf    = (unsigned short*)(smem + 65536);      // 16*1024 bf16
    float*          red    = (float*)(smem + 65536 + 32768);       // reductions

    const int tid  = threadIdx.x;      // 128 threads, 4 waves
    const int lane = tid & 31;
    const int wave = tid >> 5;
    const int l15  = lane & 15;
    const int hsel = (lane < 16) ? 0 : 1;

    const int wg = blockIdx.x;                  // b*1024 + h*64 + qt
    const int qt = wg & 63;
    const int h  = (wg >> 6) & 15;
    const int b  = wg >> 10;
    const int qrow0 = qt * 16;
    const size_t tokbase = (size_t)b * 1024;
    const float scale = 0.125f;                 // 1/sqrt(64)

    // ---- load Q fragments once (A matrix, K dim = head_dim 0..63) ----
    FragB16 aQ[2];
    #pragma unroll
    for (int s2 = 0; s2 < 2; ++s2) {
        #pragma unroll
        for (int v = 0; v < 8; ++v) {
            const int d = s2 * 32 + ((v < 4) ? 0 : 16) + hsel * 8 + 2 * (v & 3);
            aQ[s2].u[v] = *(const unsigned int*)
                &qkv[(tokbase + qrow0 + l15) * 3072 + h * 64 + d];
        }
    }

    // ---- phase 1: scores = Q K^T * scale + bias, into LDS ----
    for (int j = wave; j < 64; j += 4) {
        AccF32 acc; acc.v = (f32x8){};
        #pragma unroll
        for (int s2 = 0; s2 < 2; ++s2) {
            FragB16 bK;   // B matrix [d][k_token]
            #pragma unroll
            for (int v = 0; v < 8; ++v) {
                const int d = s2 * 32 + ((v < 4) ? 0 : 16) + hsel * 8 + 2 * (v & 3);
                bK.u[v] = *(const unsigned int*)
                    &qkv[(tokbase + j * 16 + l15) * 3072 + 1024 + h * 64 + d];
            }
            acc.v = __builtin_amdgcn_wmma_f32_16x16x32_bf16(
                false, aQ[s2].v, false, bK.v, (short)0, acc.v, false, false);
        }
        const int col = j * 16 + l15;
        const float* brow = attn_bias
            + ((((size_t)b * 16 + h) * 1024) + qrow0) * 1024 + col;
        #pragma unroll
        for (int p = 0; p < 8; ++p) {
            const int r = p + hsel * 8;
            scores[r * 1024 + col] = acc.f[p] * scale + brow[(size_t)r * 1024];
        }
    }
    __syncthreads();

    // ---- phase 2: exact softmax over each of 16 rows (128 cols/thread) ----
    const int row  = tid & 15;
    const int part = tid >> 4;        // 8 parts
    float mx = -3.4e38f;
    for (int c = part * 128; c < part * 128 + 128; ++c)
        mx = fmaxf(mx, scores[row * 1024 + c]);
    red[part * 16 + row] = mx;
    __syncthreads();
    if (tid < 16) {
        float m = red[tid];
        for (int p = 1; p < 8; ++p) m = fmaxf(m, red[p * 16 + tid]);
        red[128 + tid] = m;
    }
    __syncthreads();
    const float rmax = red[128 + row];
    float sum = 0.0f;
    for (int c = part * 128; c < part * 128 + 128; ++c)
        sum += expf(scores[row * 1024 + c] - rmax);
    __syncthreads();
    red[part * 16 + row] = sum;
    __syncthreads();
    if (tid < 16) {
        float s = red[tid];
        for (int p = 1; p < 8; ++p) s += red[p * 16 + tid];
        red[144 + tid] = s;
    }
    __syncthreads();
    const float inv = 1.0f / red[144 + row];
    for (int c = part * 128; c < part * 128 + 128; ++c)
        pbf[row * 1024 + c] = f2bf(expf(scores[row * 1024 + c] - rmax) * inv);
    __syncthreads();

    // ---- phase 3: ctx(16x64) = P(16x1024) @ V(1024x64); wave -> 16-col block
    // V fragments come from vT[d][token] so K(token) pairs are contiguous. ----
    const int d = wave * 16 + l15;
    const unsigned short* vrow = vT + (size_t)(h * 64 + d) * 2048 + tokbase;
    AccF32 co; co.v = (f32x8){};
    for (int ks = 0; ks < 1024; ks += 32) {
        FragB16 aP, bV;
        #pragma unroll
        for (int v = 0; v < 8; ++v) {
            const int kk = ks + ((v < 4) ? 0 : 16) + hsel * 8 + 2 * (v & 3);
            aP.u[v] = *(const unsigned int*)&pbf[l15 * 1024 + kk];
        }
        #pragma unroll
        for (int v = 0; v < 8; ++v) {
            const int kt = ks + ((v < 4) ? 0 : 16) + hsel * 8 + 2 * (v & 3);
            bV.u[v] = *(const unsigned int*)&vrow[kt];
        }
        co.v = __builtin_amdgcn_wmma_f32_16x16x32_bf16(
            false, aP.v, false, bV.v, (short)0, co.v, false, false);
    }
    const int dcol = h * 64 + wave * 16 + l15;
    #pragma unroll
    for (int p = 0; p < 8; ++p) {
        const int r = p + hsel * 8;
        ctx[(tokbase + qrow0 + r) * 1024 + dcol] = f2bf(co.f[p]);
    }
}

// ---------------------------------------------------------------------------
// Host-side launch
// ---------------------------------------------------------------------------
extern "C" void kernel_launch(void* const* d_in, const int* in_sizes, int n_in,
                              void* d_out, int out_size, void* d_ws, size_t ws_size,
                              hipStream_t stream) {
    (void)in_sizes; (void)n_in; (void)out_size; (void)ws_size;

    const float* x      = (const float*)d_in[0];
    const float* abias  = (const float*)d_in[1];
    const float* ln1_g  = (const float*)d_in[2];
    const float* ln1_b  = (const float*)d_in[3];
    const float* Wqkv   = (const float*)d_in[4];
    const float* bqkv   = (const float*)d_in[5];
    const float* Wo     = (const float*)d_in[6];
    const float* bo     = (const float*)d_in[7];
    const float* ln2_g  = (const float*)d_in[8];
    const float* ln2_b  = (const float*)d_in[9];
    const float* W1     = (const float*)d_in[10];
    const float* b1     = (const float*)d_in[11];
    const float* W2     = (const float*)d_in[12];
    const float* b2     = (const float*)d_in[13];
    float* out = (float*)d_out;

    char* ws = (char*)d_ws;
    unsigned short* wqT = (unsigned short*)(ws + 0);            // [3072][1024]
    unsigned short* woT = (unsigned short*)(ws + 6291456);      // [1024][1024]
    unsigned short* w1T = (unsigned short*)(ws + 8388608);      // [4096][1024]
    unsigned short* w2T = (unsigned short*)(ws + 16777216);     // [1024][4096]
    unsigned short* hbf = (unsigned short*)(ws + 25165824);     // [2048][1024]
    unsigned short* qkv = (unsigned short*)(ws + 29360128);     // [2048][3072]
    unsigned short* ctx = (unsigned short*)(ws + 41943040);     // [2048][1024]
    float*          xres= (float*)        (ws + 46137344);      // [2048][1024] f32
    unsigned short* h2  = (unsigned short*)(ws + 54525952);     // [2048][1024]
    unsigned short* gbf = (unsigned short*)(ws + 58720256);     // [2048][4096]
    unsigned short* vT  = (unsigned short*)(ws + 75497472);     // [1024][2048]

    const int T = 2048;   // B*S tokens

    // 1) weight conversion fp32 -> bf16, transposed to [n][k]
    cvt_transpose_bf16<<<dim3(3072/32, 1024/32), 256, 0, stream>>>(Wqkv, wqT, 1024, 3072);
    cvt_transpose_bf16<<<dim3(1024/32, 1024/32), 256, 0, stream>>>(Wo,   woT, 1024, 1024);
    cvt_transpose_bf16<<<dim3(4096/32, 1024/32), 256, 0, stream>>>(W1,   w1T, 1024, 4096);
    cvt_transpose_bf16<<<dim3(1024/32, 4096/32), 256, 0, stream>>>(W2,   w2T, 4096, 1024);

    // 2) LN1
    layernorm_bf16<<<T, 256, 0, stream>>>(x, ln1_g, ln1_b, hbf);

    // 3) QKV projection (also emits transposed V into vT)
    gemm_bf16_wmma<<<dim3(3072/64, T/64), 128, 0, stream>>>(
        hbf, wqT, bqkv, nullptr, qkv, nullptr, vT, T, 3072, 1024, 0);

    // 4) attention (2*16*64 workgroups, 96KB+2KB dynamic LDS)
    attention_wmma<<<2048, 128, 65536 + 32768 + 2048, stream>>>(qkv, vT, abias, ctx);

    // 5) output projection + residual: xres = x + ctx @ Wo + bo
    gemm_bf16_wmma<<<dim3(1024/64, T/64), 128, 0, stream>>>(
        ctx, woT, bo, x, nullptr, xres, nullptr, T, 1024, 1024, 0);

    // 6) LN2
    layernorm_bf16<<<T, 256, 0, stream>>>(xres, ln2_g, ln2_b, h2);

    // 7) FFN up + mish
    gemm_bf16_wmma<<<dim3(4096/64, T/64), 128, 0, stream>>>(
        h2, w1T, b1, nullptr, gbf, nullptr, nullptr, T, 4096, 1024, 1);

    // 8) FFN down + residual: out = xres + g @ W2 + b2
    gemm_bf16_wmma<<<dim3(1024/64, T/64), 128, 0, stream>>>(
        gbf, w2T, b2, xres, nullptr, out, nullptr, T, 1024, 4096, 0);
}
